// AKIRALayer_56762287784375
// MI455X (gfx1250) — matrix operand order repaired and checked by reading, not compiled, not run on hardware
//
#include <hip/hip_runtime.h>
#include <math.h>

#define B_   8
#define T_   2048
#define D_   512
#define DB_  64
#define BT_  (B_ * T_)
#define KCONV 15
#define PADT (T_ + KCONV - 1)   /* 2062 */

#define BM 128
#define BN 64
#define BK 32

typedef __attribute__((ext_vector_type(16))) __bf16 v16bf;
typedef __attribute__((ext_vector_type(8)))  __bf16 v8bf;
typedef __attribute__((ext_vector_type(8)))  float  v8f;

/* ---------- bf16 helpers (RNE) ---------- */
__device__ __forceinline__ __bf16 f2bf(float f) {
  union { float f; unsigned u; } a; a.f = f;
  unsigned r = a.u + 0x7FFFu + ((a.u >> 16) & 1u);
  unsigned short s = (unsigned short)(r >> 16);
  union { unsigned short s; __bf16 h; } c; c.s = s; return c.h;
}
__device__ __forceinline__ float bf2f(__bf16 h) {
  union { unsigned short s; __bf16 h; } c; c.h = h;
  union { unsigned u; float f; } a; a.u = ((unsigned)c.s) << 16; return a.f;
}

/* ---------- block-tiled WMMA GEMM: C = A(bf16) * B(bf16) + bias ----------
 * 256 threads = 8 waves. Block tile 128x64, k-step 32. Wave w computes rows
 * [w*16, w*16+16) x all 64 cols: 4 accumulators, A fragment reused 4x.
 * LDS: A row-major 128x32; B stored TRANSPOSED [col][k] so each lane's WMMA
 * B-fragment (bv[v] = B[half*16+v][col]) is one contiguous 32-byte LDS read.
 * Requires M%128==0, N%64==0, K%32==0 (true for all call sites).
 * A/B fragments use contiguous k = half*16 + s on both sides; C/D map per CDNA5 ISA 7.12.2. */
__launch_bounds__(256)
__global__ void wmma_gemm_tiled(const __bf16* __restrict__ A, int lda, size_t strideA,
                                const __bf16* __restrict__ Bw, int ldb,
                                const float* __restrict__ bias,
                                float* __restrict__ Cf, __bf16* __restrict__ Cb,
                                int ldc, size_t strideC, int M, int N, int K)
{
  __shared__ __bf16 lA[BM * BK];      /* 8 KB, row-major */
  __shared__ __bf16 lBt[BN * BK];     /* 4 KB, [col][k]  */
  const int tid  = threadIdx.x;
  const int wave = tid >> 5;
  const int lane = tid & 31;
  const int half = lane >> 4;
  const int l16  = lane & 15;
  const int nblk = N / BN;
  const int m0 = (blockIdx.x / nblk) * BM;
  const int n0 = (blockIdx.x % nblk) * BN;
  const size_t abase = (size_t)blockIdx.y * strideA;
  const size_t cbase = (size_t)blockIdx.y * strideC;

  /* staging assignments (exact, no divergence) */
  const int ar   = tid >> 2;          /* A row 0..63 (and +64)      */
  const int aq   = (tid & 3) * 8;     /* A 8-col group within BK    */
  const int bkk  = tid >> 3;          /* B k-row 0..31              */
  const int bcol = (tid & 7) * 8;     /* B 8-col group within BN    */

  v8f acc[4] = {};

  for (int k0 = 0; k0 < K; k0 += BK) {
    __syncthreads();
    /* ---- stage A: two coalesced 16B loads per thread ---- */
    {
      const __bf16* g0 = A + abase + (size_t)(m0 + ar) * lda + k0 + aq;
      const __bf16* g1 = A + abase + (size_t)(m0 + ar + 64) * lda + k0 + aq;
      __builtin_prefetch(g0 + BK, 0, 1);
      *(v8bf*)&lA[ar * BK + aq]        = *(const v8bf*)g0;
      *(v8bf*)&lA[(ar + 64) * BK + aq] = *(const v8bf*)g1;
    }
    /* ---- stage B transposed: one 16B load + 8 u16 LDS scatters ---- */
    {
      const v8bf bvec = *(const v8bf*)(Bw + (size_t)(k0 + bkk) * ldb + n0 + bcol);
#pragma unroll
      for (int e = 0; e < 8; ++e)
        lBt[(bcol + e) * BK + bkk] = bvec[e];
    }
    __syncthreads();
    /* ---- A fragment: two contiguous ds_load_b128 ---- */
    const int arow = wave * 16 + l16;
    const v8bf alo = *(const v8bf*)&lA[arow * BK + half * 16];
    const v8bf ahi = *(const v8bf*)&lA[arow * BK + half * 16 + 8];
    const v16bf av = __builtin_shufflevector(alo, ahi,
        0, 1, 2, 3, 4, 5, 6, 7, 8, 9, 10, 11, 12, 13, 14, 15);
    /* ---- 4 WMMAs, B fragments contiguous 32B LDS reads ---- */
#pragma unroll
    for (int j = 0; j < 4; ++j) {
      const v16bf bv = *(const v16bf*)&lBt[(j * 16 + l16) * BK + half * 16];
      acc[j] = __builtin_amdgcn_wmma_f32_16x16x32_bf16(false, av, false, bv,
                                                       (short)0, acc[j], false, false);
    }
  }

  /* ---- epilogue: C/D map lane->(col = l16, rows r+8*half) ---- */
#pragma unroll
  for (int j = 0; j < 4; ++j) {
    const int col = n0 + j * 16 + l16;
    const float bb = bias ? bias[col] : 0.0f;
#pragma unroll
    for (int r = 0; r < 8; ++r) {
      const int rowc = m0 + wave * 16 + r + 8 * half;
      const float v = acc[j][r] + bb;
      if (Cf) Cf[cbase + (size_t)rowc * ldc + col] = v;
      else    Cb[cbase + (size_t)rowc * ldc + col] = f2bf(v);
    }
  }
}

/* ---------- naive fp32 GEMM for the tiny DB-sized projections ---------- */
__global__ void naive_gemm(const float* __restrict__ A, int lda,
                           const float* __restrict__ Bw, int ldb,
                           const float* __restrict__ bias,
                           float* __restrict__ Cf, __bf16* __restrict__ Cb, int ldc,
                           long M, int N, int K)
{
  const long total = M * (long)N;
  for (long idx = (long)blockIdx.x * blockDim.x + threadIdx.x; idx < total;
       idx += (long)gridDim.x * blockDim.x) {
    const long m = idx / N;
    const int  n = (int)(idx % N);
    float acc = bias ? bias[n] : 0.0f;
    const float* a = A + m * (long)lda;
    for (int k = 0; k < K; ++k) acc += a[k] * Bw[(size_t)k * ldb + n];
    if (Cf) Cf[m * (long)ldc + n] = acc;
    else    Cb[m * (long)ldc + n] = f2bf(acc);
  }
}

/* ---------- elementwise / reshape helpers ---------- */
__global__ void f32_to_bf16_k(const float* __restrict__ in, __bf16* __restrict__ out, size_t n) {
  size_t i = (size_t)blockIdx.x * blockDim.x + threadIdx.x;
  if (i < n) out[i] = f2bf(in[i]);
}
/* dec_conv_w [o=512][i=512][k=15] -> Bmat [(k*512+i)][o] bf16 */
__global__ void conv_w_reshape_k(const float* __restrict__ w, __bf16* __restrict__ out) {
  size_t idx = (size_t)blockIdx.x * blockDim.x + threadIdx.x;
  const size_t n = (size_t)D_ * D_ * KCONV;
  if (idx >= n) return;
  const int o = (int)(idx / (D_ * KCONV));
  const int r = (int)(idx % (D_ * KCONV));
  const int i = r / KCONV;
  const int k = r % KCONV;
  out[((size_t)k * D_ + i) * D_ + o] = f2bf(w[idx]);
}
/* h fp32 [B,T,D] -> hpad bf16 [B,14+T,D] with 14 zero rows of left padding */
__global__ void pad_convert_k(const float* __restrict__ h, __bf16* __restrict__ hpad, size_t n) {
  size_t i = (size_t)blockIdx.x * blockDim.x + threadIdx.x;
  if (i >= n) return;
  const size_t per = (size_t)PADT * D_;
  const int b = (int)(i / per);
  const size_t rem = i % per;
  const int r = (int)(rem / D_);
  const int c = (int)(rem % D_);
  float v = 0.0f;
  if (r >= KCONV - 1)
    v = h[((size_t)b * T_ + (r - (KCONV - 1))) * D_ + c];
  hpad[i] = f2bf(v);
}
/* y_gated_bf16 = bf16( y * sigmoid(gate[ch]) )  (conv bias already added) */
__global__ void gate_bf16_k(const float* __restrict__ y, const float* __restrict__ gate,
                            __bf16* __restrict__ out, size_t n) {
  size_t i = (size_t)blockIdx.x * blockDim.x + threadIdx.x;
  if (i >= n) return;
  const int ch = (int)(i & (D_ - 1));
  const float g = 1.0f / (1.0f + expf(-gate[ch]));
  out[i] = f2bf(y[i] * g);
}
__global__ void residual_k(const float* __restrict__ x, const float* __restrict__ c,
                           float* __restrict__ out, size_t n) {
  size_t i = (size_t)blockIdx.x * blockDim.x + threadIdx.x;
  if (i < n) out[i] = x[i] + c[i];
}

/* ---------- per-band processor: LN + gelu MLP (+ optional sigmoid gate) ---------- */
__launch_bounds__(64)
__global__ void band_mlp(float* __restrict__ zb,
                         const float* __restrict__ ng, const float* __restrict__ nb,
                         const float* __restrict__ fc1w, const float* __restrict__ fc1b,
                         const float* __restrict__ fc2w, const float* __restrict__ fc2b,
                         const float* __restrict__ gw, const float* __restrict__ gb,
                         int H, int use_gate)
{
  __shared__ float z[DB_], zn[DB_], hid[256], red[DB_];
  const int tid = threadIdx.x;
  const long row = blockIdx.x;
  float* zr = zb + row * DB_;
  z[tid] = zr[tid];
  __syncthreads();
  red[tid] = z[tid]; __syncthreads();
  for (int s = 32; s > 0; s >>= 1) { if (tid < s) red[tid] += red[tid + s]; __syncthreads(); }
  const float mean = red[0] * (1.0f / DB_);
  __syncthreads();
  const float d = z[tid] - mean;
  red[tid] = d * d; __syncthreads();
  for (int s = 32; s > 0; s >>= 1) { if (tid < s) red[tid] += red[tid + s]; __syncthreads(); }
  const float inv = rsqrtf(red[0] * (1.0f / DB_) + 1e-5f);
  zn[tid] = (z[tid] - mean) * inv * ng[tid] + nb[tid];
  __syncthreads();
  for (int h = tid; h < H; h += DB_) {
    float a = fc1b[h];
    for (int k = 0; k < DB_; ++k) a += zn[k] * fc1w[k * H + h];
    hid[h] = 0.5f * a * (1.0f + erff(a * 0.70710678118654752f));  /* exact gelu */
  }
  __syncthreads();
  float o = fc2b[tid];
  for (int k = 0; k < H; ++k) o += hid[k] * fc2w[k * DB_ + tid];
  float out;
  if (use_gate) {
    float g = gb[tid];
    for (int k = 0; k < DB_; ++k) g += z[k] * gw[k * DB_ + tid];
    g = 1.0f / (1.0f + expf(-g));
    out = z[tid] + g * o;
  } else {
    out = z[tid] + o;
  }
  zr[tid] = out;
}

/* ---------- decay-weighted history attention; out = attn@V + zb ---------- */
__launch_bounds__(128)
__global__ void attn_kernel(const float* __restrict__ qp,
                            const __bf16* __restrict__ kp,
                            const __bf16* __restrict__ vp,
                            const float* __restrict__ zb,
                            float* __restrict__ outb, int Th)
{
  __shared__ float q[DB_];
  __shared__ float sc[128];
  __shared__ float red[128];
  const int tid = threadIdx.x;
  const long row = blockIdx.x;          /* b*T + t */
  const int t = (int)(row % T_);
  if (tid < DB_) q[tid] = qp[row * DB_ + tid];
  __syncthreads();
  float s = -3.0e38f;
  if (tid < Th) {
    const __bf16* kr = kp + ((size_t)t * Th + tid) * DB_;
    float a = 0.0f;
    for (int d = 0; d < DB_; ++d) a += q[d] * bf2f(kr[d]);
    a *= 0.125f;                         /* 1/sqrt(64) */
    a += logf(powf(0.95f, (float)(Th - 1 - tid)) + 1e-10f);
    s = a;
  }
  sc[tid] = s; red[tid] = s;
  __syncthreads();
  for (int st = 64; st > 0; st >>= 1) { if (tid < st) red[tid] = fmaxf(red[tid], red[tid + st]); __syncthreads(); }
  const float mx = red[0];
  __syncthreads();
  const float e = (tid < Th) ? expf(sc[tid] - mx) : 0.0f;
  red[tid] = e;
  __syncthreads();
  for (int st = 64; st > 0; st >>= 1) { if (tid < st) red[tid] += red[tid + st]; __syncthreads(); }
  const float denom = red[0];
  __syncthreads();
  sc[tid] = e / denom;
  __syncthreads();
  if (tid < DB_) {
    float a = 0.0f;
    for (int j = 0; j < Th; ++j)
      a += sc[j] * bf2f(vp[((size_t)t * Th + j) * DB_ + tid]);
    outb[row * DB_ + tid] = a + zb[row * DB_ + tid];
  }
}

/* ---------- host side ---------- */
static inline long cdivl(long a, long b) { return (a + b - 1) / b; }

static void run_wmma(const __bf16* A, int lda, size_t sA,
                     const __bf16* Bw, int ldb, const float* bias,
                     float* Cf, __bf16* Cb, int ldc, size_t sC,
                     int M, int N, int K, int batches, hipStream_t stream)
{
  dim3 grid((unsigned)((M / BM) * (N / BN)), (unsigned)batches);
  wmma_gemm_tiled<<<grid, 256, 0, stream>>>(A, lda, sA, Bw, ldb, bias, Cf, Cb, ldc, sC, M, N, K);
}

extern "C" void kernel_launch(void* const* d_in, const int* in_sizes, int n_in,
                              void* d_out, int out_size, void* d_ws, size_t ws_size,
                              hipStream_t stream)
{
  if (n_in < 47) return;
  const float* x          = (const float*)d_in[0];
  const float* dec_in_w   = (const float*)d_in[1];
  const float* dec_in_b   = (const float*)d_in[2];
  const float* dec_conv_w = (const float*)d_in[3];
  const float* dec_conv_b = (const float*)d_in[4];
  const float* dec_gate   = (const float*)d_in[5];
  const float* dec_out_w  = (const float*)d_in[6];
  const float* dec_out_b  = (const float*)d_in[7];
  const float* geo_ng  = (const float*)d_in[8];   const float* geo_nb  = (const float*)d_in[9];
  const float* geo_f1w = (const float*)d_in[10];  const float* geo_f1b = (const float*)d_in[11];
  const float* geo_f2w = (const float*)d_in[12];  const float* geo_f2b = (const float*)d_in[13];
  const float* geo_gw  = (const float*)d_in[14];  const float* geo_gb  = (const float*)d_in[15];
  const float* hyb_ng  = (const float*)d_in[16];  const float* hyb_nb  = (const float*)d_in[17];
  const float* hyb_f1w = (const float*)d_in[18];  const float* hyb_f1b = (const float*)d_in[19];
  const float* hyb_f2w = (const float*)d_in[20];  const float* hyb_f2b = (const float*)d_in[21];
  const float* rea_ng  = (const float*)d_in[22];  const float* rea_nb  = (const float*)d_in[23];
  const float* rea_f1w = (const float*)d_in[24];  const float* rea_f1b = (const float*)d_in[25];
  const float* rea_f2w = (const float*)d_in[26];  const float* rea_f2b = (const float*)d_in[27];
  const float* ha_qw = (const float*)d_in[28];    const float* ha_qb = (const float*)d_in[29];
  const float* ha_kw = (const float*)d_in[30];    const float* ha_kb = (const float*)d_in[31];
  const float* ha_vw = (const float*)d_in[32];    const float* ha_vb = (const float*)d_in[33];
  const float* ha_ow = (const float*)d_in[34];    const float* ha_ob = (const float*)d_in[35];
  const float* worm_w = (const float*)d_in[36];   const float* worm_b = (const float*)d_in[37];
  const float* out_w  = (const float*)d_in[38];   const float* out_b  = (const float*)d_in[39];
  const float* hist[7];
  for (int i = 0; i < 7; ++i) hist[i] = (const float*)d_in[40 + i];
  const int Th[7] = {128, 64, 32, 16, 16, 8, 4};

  /* workspace bump allocator */
  char* ws = (char*)d_ws;
  size_t off = 0;
  auto alloc = [&](size_t bytes) -> char* {
    char* p = ws + off; off += (bytes + 255) & ~(size_t)255; return p;
  };
  __bf16* xb   = (__bf16*)alloc((size_t)BT_ * D_ * 2);
  __bf16* wa   = (__bf16*)alloc((size_t)KCONV * D_ * D_ * 2);  /* conv B matrix */
  __bf16* wb   = (__bf16*)alloc((size_t)D_ * D_ * 2);          /* generic weight buf */
  __bf16* hpad = (__bf16*)alloc((size_t)B_ * PADT * D_ * 2);
  float*  fbuf = (float*) alloc((size_t)BT_ * D_ * 4);
  __bf16* ycb  = (__bf16*)alloc((size_t)BT_ * D_ * 2);
  float*  zbuf = (float*) alloc((size_t)BT_ * DB_ * 4);
  float*  qp   = (float*) alloc((size_t)BT_ * DB_ * 4);
  float*  attb = (float*) alloc((size_t)BT_ * DB_ * 4);
  __bf16* kp   = (__bf16*)alloc((size_t)T_ * 128 * DB_ * 2);
  __bf16* vp   = (__bf16*)alloc((size_t)T_ * 128 * DB_ * 2);
  __bf16* ycat = (__bf16*)alloc((size_t)BT_ * D_ * 2);
  if (off > ws_size) return;

  {
    size_t n = (size_t)BT_ * D_;
    f32_to_bf16_k<<<(unsigned)cdivl((long)n, 256), 256, 0, stream>>>(x, xb, n);
  }

  for (int b = 0; b < 7; ++b) {
    /* 1. input projection: h = x @ Win + bin  (WMMA) */
    { size_t n = (size_t)D_ * D_;
      f32_to_bf16_k<<<(unsigned)cdivl((long)n, 256), 256, 0, stream>>>(dec_in_w + (size_t)b * n, wb, n); }
    run_wmma(xb, D_, 0, wb, D_, dec_in_b + b * D_, fbuf, nullptr, D_, 0, BT_, D_, D_, 1, stream);

    /* 2. causal conv as strided GEMM, K = 15*512 (WMMA, batched over B) */
    { size_t n = (size_t)B_ * PADT * D_;
      pad_convert_k<<<(unsigned)cdivl((long)n, 256), 256, 0, stream>>>(fbuf, hpad, n); }
    { size_t n = (size_t)D_ * D_ * KCONV;
      conv_w_reshape_k<<<(unsigned)cdivl((long)n, 256), 256, 0, stream>>>(dec_conv_w + (size_t)b * n, wa); }
    run_wmma(hpad, D_, (size_t)PADT * D_, wa, D_, dec_conv_b + b * D_,
             fbuf, nullptr, D_, (size_t)T_ * D_, T_, D_, KCONV * D_, B_, stream);

    /* 3. gate, then band projection [*,512]x[512,64] (WMMA) */
    { size_t n = (size_t)BT_ * D_;
      gate_bf16_k<<<(unsigned)cdivl((long)n, 256), 256, 0, stream>>>(fbuf, dec_gate + b * D_, ycb, n); }
    { size_t n = (size_t)D_ * DB_;
      f32_to_bf16_k<<<(unsigned)cdivl((long)n, 256), 256, 0, stream>>>(dec_out_w + (size_t)b * n, wb, n); }
    run_wmma(ycb, D_, 0, wb, DB_, dec_out_b + b * DB_, zbuf, nullptr, DB_, 0, BT_, DB_, D_, 1, stream);

    /* 4. per-band processor */
    if (b < 3) {
      band_mlp<<<BT_, 64, 0, stream>>>(zbuf, geo_ng + b * DB_, geo_nb + b * DB_,
          geo_f1w + (size_t)b * DB_ * 128, geo_f1b + b * 128,
          geo_f2w + (size_t)b * 128 * DB_, geo_f2b + b * DB_,
          geo_gw + (size_t)b * DB_ * DB_, geo_gb + b * DB_, 128, 1);
    } else if (b < 5) {
      const int i = b - 3;
      band_mlp<<<BT_, 64, 0, stream>>>(zbuf, hyb_ng + i * DB_, hyb_nb + i * DB_,
          hyb_f1w + (size_t)i * DB_ * 192, hyb_f1b + i * 192,
          hyb_f2w + (size_t)i * 192 * DB_, hyb_f2b + i * DB_, nullptr, nullptr, 192, 0);
    } else {
      const int i = b - 5;
      band_mlp<<<BT_, 64, 0, stream>>>(zbuf, rea_ng + i * DB_, rea_nb + i * DB_,
          rea_f1w + (size_t)i * DB_ * 256, rea_f1b + i * 256,
          rea_f2w + (size_t)i * 256 * DB_, rea_f2b + i * DB_, nullptr, nullptr, 256, 0);
    }

    /* 5. history attention */
    { long tot = (long)BT_ * DB_;
      naive_gemm<<<(unsigned)cdivl(tot, 256), 256, 0, stream>>>(zbuf, DB_,
          ha_qw + (size_t)b * DB_ * DB_, DB_, ha_qb + b * DB_, qp, nullptr, DB_, BT_, DB_, DB_); }
    { long Mh = (long)T_ * Th[b]; long tot = Mh * DB_;
      naive_gemm<<<(unsigned)cdivl(tot, 256), 256, 0, stream>>>(hist[b], DB_,
          ha_kw + (size_t)b * DB_ * DB_, DB_, ha_kb + b * DB_, nullptr, kp, DB_, Mh, DB_, DB_);
      naive_gemm<<<(unsigned)cdivl(tot, 256), 256, 0, stream>>>(hist[b], DB_,
          ha_vw + (size_t)b * DB_ * DB_, DB_, ha_vb + b * DB_, nullptr, vp, DB_, Mh, DB_, DB_); }
    attn_kernel<<<BT_, 128, 0, stream>>>(qp, kp, vp, zbuf, attb, Th[b]);
    { long tot = (long)BT_ * DB_;
      naive_gemm<<<(unsigned)cdivl(tot, 256), 256, 0, stream>>>(attb, DB_,
          ha_ow + (size_t)b * DB_ * DB_, DB_, ha_ob + b * DB_, nullptr, ycat + b * DB_, D_, BT_, DB_, DB_); }
  }

  /* wormhole band -> columns 448..511 (WMMA) */
  { size_t n = (size_t)D_ * DB_;
    f32_to_bf16_k<<<(unsigned)cdivl((long)n, 256), 256, 0, stream>>>(worm_w, wb, n); }
  run_wmma(xb, D_, 0, wb, DB_, worm_b, nullptr, ycat + 7 * DB_, D_, 0, BT_, DB_, D_, 1, stream);

  /* final output projection + residual (WMMA) */
  { size_t n = (size_t)D_ * D_;
    f32_to_bf16_k<<<(unsigned)cdivl((long)n, 256), 256, 0, stream>>>(out_w, wb, n); }
  run_wmma(ycat, D_, 0, wb, D_, out_b, fbuf, nullptr, D_, 0, BT_, D_, D_, 1, stream);
  { size_t n = (size_t)BT_ * D_;
    residual_k<<<(unsigned)cdivl((long)n, 256), 256, 0, stream>>>(x, fbuf, (float*)d_out, n); }
}
